// DualAttention_35991825940614
// MI455X (gfx1250) — matrix operand order repaired and checked by reading, not compiled, not run on hardware
//
#include <hip/hip_runtime.h>

// ---------------------------------------------------------------------------
// DualAttention fused implementation for gfx1250 (MI455X, wave32, WMMA)
// ---------------------------------------------------------------------------

typedef __attribute__((ext_vector_type(16))) _Float16     v16h;
typedef __attribute__((ext_vector_type(4)))  _Float16     v4h;
typedef __attribute__((ext_vector_type(8)))  float        v8f;
typedef __attribute__((ext_vector_type(8)))  unsigned int v8u;

constexpr int SS = 1024;  // sequence
constexpr int BB = 64;    // batch
constexpr int DD = 512;   // encoder dim
constexpr int FF = 256;   // field dim
constexpr int HH = 512;   // hidden dim

// ---------------------------------------------------------------------------
// Pack fp32 weight matrix [KTOT, HH] into f16 WMMA-B fragment order.
// Tile (nt, kt) covers W[kt*32 : kt*32+32, nt*16 : nt*16+16]; within a tile,
// lane l (0..31) holds column (nt*16 + (l&15)), K values ((l>>4)*16 + j),
// j = 0..15, stored contiguously so the main loop loads v16h in one shot.
// ---------------------------------------------------------------------------
__global__ __launch_bounds__(256) void pack_weights(const float* __restrict__ W,
                                                    _Float16* __restrict__ out,
                                                    int KTOT) {
  int t = blockIdx.x * blockDim.x + threadIdx.x;
  int total = KTOT * HH;
  if (t >= total) return;
  int j   = t & 15;
  int l   = (t >> 4) & 31;
  int tk  = t >> 9;               // tile index, 512 halves per tile
  int NKT = KTOT >> 5;
  int kt  = tk & (NKT - 1);
  int nt  = tk / NKT;
  int k   = kt * 32 + ((l >> 4) << 4) + j;
  int n   = (nt << 4) + (l & 15);
  out[t]  = (_Float16)W[k * HH + n];
}

// ---------------------------------------------------------------------------
// state_tanh_f = tanh(x @ W_sf + b_sf), state_tanh_x = tanh(x @ W_se + b_se)
// grid (B, H/256, 2), 256 threads. Coalesced over h.
// ---------------------------------------------------------------------------
__global__ __launch_bounds__(256) void state_kernel(const float* __restrict__ x,
                                                    const float* __restrict__ W_sf,
                                                    const float* __restrict__ b_sf,
                                                    const float* __restrict__ W_se,
                                                    const float* __restrict__ b_se,
                                                    float* __restrict__ stf,
                                                    float* __restrict__ stx) {
  __shared__ float xs[DD];
  const int b = blockIdx.x;
  for (int i = threadIdx.x; i < DD; i += 256) xs[i] = x[b * DD + i];
  __syncthreads();
  const int h      = blockIdx.y * 256 + threadIdx.x;
  const int which  = blockIdx.z;
  const float* W   = which ? W_se : W_sf;
  const float* bia = which ? b_se : b_sf;
  float acc = bia[h];
  for (int d = 0; d < DD; ++d) acc += xs[d] * W[d * HH + h];
  (which ? stx : stf)[b * HH + h] = tanhf(acc);
}

// ---------------------------------------------------------------------------
// Fused score phase for one wave:
//   score[r] = sum_n tanh( (A @ W)[r,n] + bias[n] ) * st[n]
// A = 16 contiguous rows of [KTOT] fp32 (staged to f16 LDS), W pre-packed.
// WMMA: A 16x32 f16 (contiguous K = 16*half + s, this code's order),
// B 32x16 f16 (same contiguous K order as A), C/D 16x16 f32.
// ---------------------------------------------------------------------------
template <int KTOT>
__device__ __forceinline__ void score_phase(const float* __restrict__ Arows,
                                            const _Float16* __restrict__ Wpk,
                                            const float* __restrict__ bias,
                                            const float* __restrict__ st,
                                            float* __restrict__ scoreOut,
                                            _Float16* lds, int lane) {
  constexpr int NKT = KTOT / 32;

  // Stage 16 rows x KTOT fp32 -> f16 LDS strip (rows contiguous in memory).
  const float4* src4 = (const float4*)Arows;
#pragma unroll 4
  for (int i = lane; i < (16 * KTOT) / 4; i += 32) {
    float4 f = src4[i];
    v4h h = { (_Float16)f.x, (_Float16)f.y, (_Float16)f.z, (_Float16)f.w };
    *(v4h*)(lds + i * 4) = h;  // ds_store_b64
  }
  __syncthreads();

  const int m    = lane & 15;
  const int half = lane >> 4;
  float sacc[8];
#pragma unroll
  for (int v = 0; v < 8; ++v) sacc[v] = 0.0f;

  for (int nt = 0; nt < HH / 16; ++nt) {
    v8f acc = {};
    const _Float16* wt = Wpk + (size_t)(nt * NKT) * 512 + lane * 16;
    for (int kt = 0; kt < NKT; ++kt) {
      // A fragment: lane half 0 -> K {0..15}, half 1 -> K {16..31}
      v8u au;
#pragma unroll
      for (int v = 0; v < 8; ++v) {
        const int koff = (half << 4) + (v << 1);
        au[v] = *(const unsigned int*)(lds + m * KTOT + kt * 32 + koff);
      }
      v16h a = __builtin_bit_cast(v16h, au);
      // B fragment: contiguous 32 bytes from packed weights
      v16h b = *(const v16h*)(wt + (size_t)kt * 512);
      acc = __builtin_amdgcn_wmma_f32_16x16x32_f16(false, a, false, b,
                                                   (short)0, acc, false, false);
    }
    // D layout: VGPR v -> row (v + 8*half), col n = nt*16 + (lane&15)
    const int n    = (nt << 4) + m;
    const float bi = bias[n];
    const float sv = st[n];
#pragma unroll
    for (int v = 0; v < 8; ++v)
      sacc[v] += tanhf(acc[v] + bi) * sv;
  }

  // Reduce over the 16 lanes that share each row (butterfly within halves).
#pragma unroll
  for (int v = 0; v < 8; ++v) {
    float s = sacc[v];
    s += __shfl_xor(s, 1, 32);
    s += __shfl_xor(s, 2, 32);
    s += __shfl_xor(s, 4, 32);
    s += __shfl_xor(s, 8, 32);
    sacc[v] = s;
  }
  if (m == 0) {
#pragma unroll
    for (int v = 0; v < 8; ++v)
      scoreOut[half * 8 + v] = sacc[v];
  }
  __syncthreads();  // safe LDS strip reuse by the next phase
}

// ---------------------------------------------------------------------------
// Score kernel: grid = (B*S)/64 blocks, 128 threads (4 waves x 16 rows).
// Never materializes [S,B,H]; writes scores_e/scores_f [B*S] (row = b*S + s).
// ---------------------------------------------------------------------------
__global__ __launch_bounds__(128) void score_kernel(
    const float* __restrict__ enc, const float* __restrict__ fields,
    const _Float16* __restrict__ We_pk, const _Float16* __restrict__ Wf_pk,
    const float* __restrict__ b_e, const float* __restrict__ b_f,
    const float* __restrict__ stx, const float* __restrict__ stf,
    float* __restrict__ scores_e, float* __restrict__ scores_f) {
  __shared__ _Float16 Alds[4 * 16 * DD];  // 64 KB: one 16xD strip per wave
  const int tid     = threadIdx.x;
  const int w       = tid >> 5;
  const int lane    = tid & 31;
  const int rowBase = blockIdx.x * 64 + w * 16;
  const int b       = (blockIdx.x * 64) / SS;  // 64 | S, so one b per block
  _Float16* myLds   = Alds + w * (16 * DD);

  // Warm the fields strip in GL2 while the enc phase runs.
  __builtin_prefetch(fields + (size_t)rowBase * FF, 0, 1);

  score_phase<DD>(enc + (size_t)rowBase * DD, We_pk, b_e, stx + b * HH,
                  scores_e + rowBase, myLds, lane);
  score_phase<FF>(fields + (size_t)rowBase * FF, Wf_pk, b_f, stf + b * HH,
                  scores_f + rowBase, myLds, lane);
}

// ---------------------------------------------------------------------------
// Block reductions
// ---------------------------------------------------------------------------
__device__ __forceinline__ float blockReduceSum(float v, float* red) {
  const int tid = threadIdx.x;
  __syncthreads();
  red[tid] = v;
  __syncthreads();
  for (int off = 128; off > 0; off >>= 1) {
    if (tid < off) red[tid] += red[tid + off];
    __syncthreads();
  }
  return red[0];
}

__device__ __forceinline__ float blockReduceMax(float v, float* red) {
  const int tid = threadIdx.x;
  __syncthreads();
  red[tid] = v;
  __syncthreads();
  for (int off = 128; off > 0; off >>= 1) {
    if (tid < off) red[tid] = fmaxf(red[tid], red[tid + off]);
    __syncthreads();
  }
  return red[0];
}

// ---------------------------------------------------------------------------
// Dual softmax over S + gamma normalization. grid = B blocks, 256 threads.
// Writes gammas both to ws (per-b contiguous) and to d_out in [S,B] layout.
// ---------------------------------------------------------------------------
__global__ __launch_bounds__(256) void gamma_kernel(const float* __restrict__ scores_e,
                                                    const float* __restrict__ scores_f,
                                                    float* __restrict__ gam,
                                                    float* __restrict__ gam_out) {
  __shared__ float se[SS];
  __shared__ float sf[SS];
  __shared__ float red[256];
  const int b   = blockIdx.x;
  const int tid = threadIdx.x;
  for (int s = tid; s < SS; s += 256) {
    se[s] = scores_e[b * SS + s];
    sf[s] = scores_f[b * SS + s];
  }
  float me = -INFINITY, mf = -INFINITY;
  __syncthreads();
  for (int s = tid; s < SS; s += 256) {
    me = fmaxf(me, se[s]);
    mf = fmaxf(mf, sf[s]);
  }
  me = blockReduceMax(me, red);
  mf = blockReduceMax(mf, red);
  float pe = 0.f, pf = 0.f;
  for (int s = tid; s < SS; s += 256) {
    pe += __expf(se[s] - me);
    pf += __expf(sf[s] - mf);
  }
  const float sume = blockReduceSum(pe, red);
  const float sumf = blockReduceSum(pf, red);
  const float inv  = 1.0f / (sume * sumf);
  float gp = 0.f;
  for (int s = tid; s < SS; s += 256) {
    float ab = __expf(se[s] - me) * __expf(sf[s] - mf) * inv;  // alpha*beta
    se[s] = ab;  // each thread rewrites only its own slots
    gp += ab;
  }
  const float gsum  = blockReduceSum(gp, red);
  const float scale = 1.0f / (1e-6f + gsum);
  for (int s = tid; s < SS; s += 256) {
    float g = se[s] * scale;
    gam[b * SS + s]     = g;         // [b][s] for context kernel
    gam_out[s * BB + b] = g;         // reference [S,B] output layout
  }
}

// ---------------------------------------------------------------------------
// context[b,d] = sum_s gamma[b,s] * enc[b,s,d]. grid (B, D/256), 256 threads.
// enc re-read mostly hits the 192MB L2 after the score pass.
// ---------------------------------------------------------------------------
__global__ __launch_bounds__(256) void context_kernel(const float* __restrict__ enc,
                                                      const float* __restrict__ gam,
                                                      float* __restrict__ ctx) {
  __shared__ float gs[256];
  const int b = blockIdx.x;
  const int d = blockIdx.y * 256 + threadIdx.x;
  float acc = 0.f;
  for (int s0 = 0; s0 < SS; s0 += 256) {
    __syncthreads();
    gs[threadIdx.x] = gam[b * SS + s0 + threadIdx.x];
    __syncthreads();
    for (int i = 0; i < 256; ++i)
      acc += gs[i] * enc[(size_t)(b * SS + s0 + i) * DD + d];
  }
  ctx[b * DD + d] = acc;
}

// ---------------------------------------------------------------------------
// out[b,h] = tanh(ctx[b,:]@W_g[0:D] + x[b,:]@W_g[D:2D] + b_g). grid (B, H/256).
// ---------------------------------------------------------------------------
__global__ __launch_bounds__(256) void out_kernel(const float* __restrict__ ctx,
                                                  const float* __restrict__ x,
                                                  const float* __restrict__ W_g,
                                                  const float* __restrict__ b_g,
                                                  float* __restrict__ out) {
  __shared__ float cs[DD];
  __shared__ float xs[DD];
  const int b = blockIdx.x;
  const int h = blockIdx.y * 256 + threadIdx.x;
  for (int i = threadIdx.x; i < DD; i += 256) {
    cs[i] = ctx[b * DD + i];
    xs[i] = x[b * DD + i];
  }
  __syncthreads();
  float acc = b_g[h];
  for (int d = 0; d < DD; ++d) acc += cs[d] * W_g[d * HH + h];
  for (int d = 0; d < DD; ++d) acc += xs[d] * W_g[(DD + d) * HH + h];
  out[b * HH + h] = tanhf(acc);
}

// ---------------------------------------------------------------------------
// Launch
// ---------------------------------------------------------------------------
extern "C" void kernel_launch(void* const* d_in, const int* in_sizes, int n_in,
                              void* d_out, int out_size, void* d_ws, size_t ws_size,
                              hipStream_t stream) {
  (void)in_sizes; (void)n_in; (void)out_size; (void)ws_size;

  const float* x      = (const float*)d_in[0];
  const float* enc    = (const float*)d_in[1];   // [B,S,D]
  const float* fields = (const float*)d_in[2];   // [B,S,F]
  const float* W_f    = (const float*)d_in[3];
  const float* b_f    = (const float*)d_in[4];
  const float* W_e    = (const float*)d_in[5];
  const float* b_e    = (const float*)d_in[6];
  const float* W_sf   = (const float*)d_in[7];
  const float* b_sf   = (const float*)d_in[8];
  const float* W_se   = (const float*)d_in[9];
  const float* b_se   = (const float*)d_in[10];
  const float* W_g    = (const float*)d_in[11];
  const float* b_g    = (const float*)d_in[12];

  float* out     = (float*)d_out;          // [B,H]
  float* gam_out = out + BB * HH;          // [S,B]

  char*  ws  = (char*)d_ws;
  size_t off = 0;
  auto alloc = [&](size_t bytes) -> void* {
    void* p = ws + off;
    off = (off + bytes + 255) & ~(size_t)255;
    return p;
  };
  _Float16* We_pk    = (_Float16*)alloc((size_t)DD * HH * sizeof(_Float16));
  _Float16* Wf_pk    = (_Float16*)alloc((size_t)FF * HH * sizeof(_Float16));
  float*    stf      = (float*)alloc((size_t)BB * HH * sizeof(float));
  float*    stx      = (float*)alloc((size_t)BB * HH * sizeof(float));
  float*    scores_e = (float*)alloc((size_t)BB * SS * sizeof(float));
  float*    scores_f = (float*)alloc((size_t)BB * SS * sizeof(float));
  float*    gam      = (float*)alloc((size_t)BB * SS * sizeof(float));
  float*    ctx      = (float*)alloc((size_t)BB * DD * sizeof(float));

  pack_weights<<<(DD * HH + 255) / 256, 256, 0, stream>>>(W_e, We_pk, DD);
  pack_weights<<<(FF * HH + 255) / 256, 256, 0, stream>>>(W_f, Wf_pk, FF);
  state_kernel<<<dim3(BB, HH / 256, 2), 256, 0, stream>>>(x, W_sf, b_sf, W_se, b_se, stf, stx);
  score_kernel<<<(BB * SS) / 64, 128, 0, stream>>>(enc, fields, We_pk, Wf_pk,
                                                   b_e, b_f, stx, stf,
                                                   scores_e, scores_f);
  gamma_kernel<<<BB, 256, 0, stream>>>(scores_e, scores_f, gam, gam_out);
  context_kernel<<<dim3(BB, DD / 256), 256, 0, stream>>>(enc, gam, ctx);
  out_kernel<<<dim3(BB, HH / 256), 256, 0, stream>>>(ctx, x, W_g, b_g, out);
}
